// GRUModel_20023137534314
// MI455X (gfx1250) — compile-verified, matrix operand order checked by reading, not run on hardware
//
#include <hip/hip_runtime.h>

// ---------------------------------------------------------------------------
// GRU (B=64, S=512, I=256, H=1024, O=256) on gfx1250 via bf16 WMMA.
//
// Plan:
//  1) One-time converts: weights -> bf16 swizzled into WMMA B-fragment tile
//     order; x -> bf16 transposed to [S*B, I] row-major.
//  2) Big parallel GEMM: G[g][t*B+b][n] = x_t @ Wx{g} + b{g}  (bf16 store).
//  3) Persistent recurrent kernel (64 blocks, 512 waves) looping t=0..511:
//       phase A: 512 waves -> z,r gates  ([64x1024]@[1024x1024] x2, WMMA)
//       grid barrier
//       phase B: 256 waves -> candidate GEMM + tanh + h update
//       grid barrier
//  4) Head GEMM: out = h @ Wf + bf.
// ---------------------------------------------------------------------------

typedef __attribute__((ext_vector_type(16))) __bf16 v16bf;
typedef __attribute__((ext_vector_type(8)))  __bf16 v8bf;
typedef __attribute__((ext_vector_type(8)))  float  v8f;

constexpr int Bsz = 64, Ssz = 512, Isz = 256, Hsz = 1024, Osz = 256;
constexpr int SB  = Ssz * Bsz;                 // 32768 GEMM rows
constexpr size_t SBH = (size_t)SB * Hsz;

// ---- workspace layout (bytes, all 256-aligned) ----------------------------
constexpr size_t OFF_G   = 0;                          // 3*SB*H bf16 (gate preacts)
constexpr size_t SZ_G    = 3ull * SBH * 2;
constexpr size_t OFF_XT  = OFF_G + SZ_G;               // SB*I bf16
constexpr size_t SZ_XT   = (size_t)SB * Isz * 2;
constexpr size_t SZ_WX   = (size_t)Isz * Hsz * 2;
constexpr size_t OFF_WXZ = OFF_XT + SZ_XT;
constexpr size_t OFF_WXR = OFF_WXZ + SZ_WX;
constexpr size_t OFF_WXH = OFF_WXR + SZ_WX;
constexpr size_t SZ_WH   = (size_t)Hsz * Hsz * 2;
constexpr size_t OFF_WHZ = OFF_WXH + SZ_WX;
constexpr size_t OFF_WHR = OFF_WHZ + SZ_WH;
constexpr size_t OFF_WHH = OFF_WHR + SZ_WH;
constexpr size_t SZ_WF   = (size_t)Hsz * Osz * 2;
constexpr size_t OFF_WF  = OFF_WHH + SZ_WH;
constexpr size_t OFF_H   = OFF_WF + SZ_WF;             // 64*1024 f32
constexpr size_t OFF_HBF = OFF_H   + (size_t)Bsz * Hsz * 4;   // bf16 mirror
constexpr size_t OFF_Z   = OFF_HBF + (size_t)Bsz * Hsz * 2;   // z gate f32
constexpr size_t OFF_RH  = OFF_Z   + (size_t)Bsz * Hsz * 4;   // r*h bf16
constexpr size_t OFF_BAR = OFF_RH  + (size_t)Bsz * Hsz * 2;   // barrier u32

// ---- WMMA fragment helpers -------------------------------------------------
// A (16x32 bf16, row-major source, stride lda):
//   lane 0-15 : row M=lane,   K = 0..7 (v0-3) and 16..23 (v4-7)
//   lane 16-31: row M=lane-16,K = 8..15        and 24..31
__device__ __forceinline__ v16bf load_afrag(const __bf16* __restrict__ A, int lda) {
  const int lane = threadIdx.x & 31;
  const __bf16* p = A + (size_t)(lane & 15) * lda + ((lane >> 4) << 3);
  v8bf lo = *(const v8bf*)(p);
  v8bf hi = *(const v8bf*)(p + 16);
  v16bf r;
#pragma unroll
  for (int j = 0; j < 8; ++j) { r[j] = lo[j]; r[j + 8] = hi[j]; }
  return r;
}

// B fragments are pre-swizzled: each 16x16x32 tile = 512 bf16, lane-major,
// so a lane grabs its 16 values with one contiguous 32-byte load.
__device__ __forceinline__ v16bf load_bfrag(const __bf16* __restrict__ T) {
  return *(const v16bf*)(T + (size_t)(threadIdx.x & 31) * 16);
}

__device__ __forceinline__ v8f wmma_bf16(v16bf a, v16bf b, v8f c) {
  return __builtin_amdgcn_wmma_f32_16x16x32_bf16(false, a, false, b,
                                                 (short)0, c, false, false);
}

__device__ __forceinline__ float sigmoidf_(float x) {
  return 1.0f / (1.0f + __expf(-x));
}

// Monotonic grid barrier (all 64 blocks resident on MI455X).
__device__ __forceinline__ void grid_sync(unsigned* bar, unsigned nblocks) {
  __threadfence();
  __syncthreads();
  if (threadIdx.x == 0) {
    unsigned ticket = __hip_atomic_fetch_add(bar, 1u, __ATOMIC_ACQ_REL,
                                             __HIP_MEMORY_SCOPE_AGENT);
    unsigned target = (ticket / nblocks + 1u) * nblocks;
    while (__hip_atomic_load(bar, __ATOMIC_ACQUIRE,
                             __HIP_MEMORY_SCOPE_AGENT) < target) {
      __builtin_amdgcn_s_sleep(1);
    }
  }
  __syncthreads();
}

// ---- one-time conversion kernels ------------------------------------------
// Swizzle row-major fp32 W[K][N] into bf16 WMMA-B tile order.
// Tile index = nt*Ktiles + kt (K-loop streams contiguously per output tile).
__global__ void k_swizzle_w(const float* __restrict__ W, __bf16* __restrict__ out,
                            int K, int N) {
  int e = blockIdx.x * blockDim.x + threadIdx.x;
  if (e >= K * N) return;
  const int ktiles = K >> 5;
  int tIdx = e >> 9, within = e & 511;
  int lane = within >> 4, j = within & 15;
  int nt = tIdx / ktiles, kt = tIdx - nt * ktiles;
  int n  = (nt << 4) + (lane & 15);
  int kb = (lane >> 4) << 3;
  int kl = (j < 8) ? (kb + j) : (16 + kb + (j - 8));
  int k  = (kt << 5) + kl;
  out[e] = (__bf16)W[(size_t)k * N + n];
}

// x[B][S][I] fp32 -> xT[t*B+b][I] bf16
__global__ void k_convert_x(const float* __restrict__ x, __bf16* __restrict__ xT) {
  int e = blockIdx.x * blockDim.x + threadIdx.x;
  if (e >= SB * Isz) return;
  int i = e & (Isz - 1);
  int row = e >> 8;          // t*B + b
  int t = row >> 6, b = row & 63;
  xT[e] = (__bf16)x[((size_t)b * Ssz + t) * Isz + i];
}

__global__ void k_zero(float* __restrict__ h, __bf16* __restrict__ hbf,
                       unsigned* __restrict__ bar) {
  int i = blockIdx.x * blockDim.x + threadIdx.x;
  if (i < Bsz * Hsz) { h[i] = 0.0f; hbf[i] = (__bf16)0.0f; }
  if (i == 0) *bar = 0u;
}

// ---- phase 1: precompute input projections for all timesteps ---------------
// jobs = 3 gates * (SB/16) Mtiles * (H/16) Ntiles, one wave each.
__global__ void k_precompute(const __bf16* __restrict__ xT,
                             const __bf16* __restrict__ wxz,
                             const __bf16* __restrict__ wxr,
                             const __bf16* __restrict__ wxh,
                             const float* __restrict__ bz,
                             const float* __restrict__ br,
                             const float* __restrict__ bh,
                             __bf16* __restrict__ G) {
  const int wave = blockIdx.x * (blockDim.x >> 5) + (threadIdx.x >> 5);
  const int MT = SB / 16, NT = Hsz / 16;           // 2048, 64
  int g   = wave / (MT * NT);
  int rem = wave - g * (MT * NT);
  int mt  = rem / NT;
  int nt  = rem - mt * NT;
  const __bf16* Wsw  = (g == 0) ? wxz : (g == 1) ? wxr : wxh;
  const float*  bias = (g == 0) ? bz  : (g == 1) ? br  : bh;
  constexpr int KT = Isz / 32;                     // 8
  v8f acc = {};
  const __bf16* Arow  = xT  + (size_t)mt * 16 * Isz;
  const __bf16* Btile = Wsw + (size_t)nt * KT * 512;
#pragma unroll
  for (int kt = 0; kt < KT; ++kt) {
    if (kt + 2 < KT) __builtin_prefetch((const void*)(Btile + (kt + 2) * 512), 0, 1);
    v16bf a = load_afrag(Arow + kt * 32, Isz);
    v16bf b = load_bfrag(Btile + (size_t)kt * 512);
    acc = wmma_bf16(a, b, acc);
  }
  const int lane = threadIdx.x & 31;
  const int nl = lane & 15, mh = (lane >> 4) << 3;
  const int n = nt * 16 + nl;
  const float bv = bias[n];
  __bf16* Gd = G + (size_t)g * SBH;
#pragma unroll
  for (int v = 0; v < 8; ++v) {
    int m = mt * 16 + mh + v;
    Gd[(size_t)m * Hsz + n] = (__bf16)(acc[v] + bv);
  }
}

// ---- phase 2: persistent recurrent kernel ----------------------------------
__global__ void __launch_bounds__(256, 1)
k_recurrent(const __bf16* __restrict__ G,
            const __bf16* __restrict__ whz,
            const __bf16* __restrict__ whr,
            const __bf16* __restrict__ whh,
            float* __restrict__ h,
            __bf16* __restrict__ hbf,
            float* __restrict__ zbuf,
            __bf16* __restrict__ rhbf,
            unsigned* __restrict__ bar,
            unsigned nblocks) {
  const int wave = blockIdx.x * 8 + (threadIdx.x >> 5);   // 0..511
  const int lane = threadIdx.x & 31;
  const int nl = lane & 15, mh = (lane >> 4) << 3;
  constexpr int KT = Hsz / 32;                            // 32 k-steps

  for (int t = 0; t < Ssz; ++t) {
    const size_t rowbase = (size_t)t * Bsz;

    { // ---- phase A: z (waves 0-255) and r (waves 256-511) gates ----
      const int gate = wave >> 8;
      const int rem  = wave & 255;
      const int mt   = rem >> 6;          // 0..3
      const int nt   = rem & 63;          // 0..63
      const __bf16* Wsw = gate ? whr : whz;
      const __bf16* Gg  = G + (size_t)gate * SBH;
      const int n = nt * 16 + nl;
      v8f acc;
#pragma unroll
      for (int v = 0; v < 8; ++v)
        acc[v] = (float)Gg[(rowbase + mt * 16 + mh + v) * Hsz + n];
      const __bf16* Arow  = hbf + (size_t)mt * 16 * Hsz;
      const __bf16* Btile = Wsw + (size_t)nt * KT * 512;
#pragma unroll 8
      for (int kt = 0; kt < KT; ++kt) {
        if (kt + 2 < KT) __builtin_prefetch((const void*)(Btile + (kt + 2) * 512), 0, 1);
        v16bf a = load_afrag(Arow + kt * 32, Hsz);
        v16bf b = load_bfrag(Btile + (size_t)kt * 512);
        acc = wmma_bf16(a, b, acc);
      }
#pragma unroll
      for (int v = 0; v < 8; ++v) {
        const size_t idx = (size_t)(mt * 16 + mh + v) * Hsz + n;
        float gv = sigmoidf_(acc[v]);
        if (gate == 0) zbuf[idx] = gv;                      // z gate
        else           rhbf[idx] = (__bf16)(gv * h[idx]);   // r*h (bf16 A)
      }
    }
    grid_sync(bar, nblocks);

    if (wave < 256) { // ---- phase B: candidate GEMM + h update ----
      const int mt = wave >> 6, nt = wave & 63;
      const int n  = nt * 16 + nl;
      const __bf16* Gg = G + 2 * SBH;
      v8f acc;
#pragma unroll
      for (int v = 0; v < 8; ++v)
        acc[v] = (float)Gg[(rowbase + mt * 16 + mh + v) * Hsz + n];
      const __bf16* Arow  = rhbf + (size_t)mt * 16 * Hsz;
      const __bf16* Btile = whh  + (size_t)nt * KT * 512;
#pragma unroll 8
      for (int kt = 0; kt < KT; ++kt) {
        if (kt + 2 < KT) __builtin_prefetch((const void*)(Btile + (kt + 2) * 512), 0, 1);
        v16bf a = load_afrag(Arow + kt * 32, Hsz);
        v16bf b = load_bfrag(Btile + (size_t)kt * 512);
        acc = wmma_bf16(a, b, acc);
      }
#pragma unroll
      for (int v = 0; v < 8; ++v) {
        const size_t idx = (size_t)(mt * 16 + mh + v) * Hsz + n;
        float nv = tanhf(acc[v]);
        float z  = zbuf[idx];
        float hn = (1.0f - z) * h[idx] + z * nv;
        h[idx]   = hn;
        hbf[idx] = (__bf16)hn;
      }
    }
    grid_sync(bar, nblocks);
  }
}

// ---- phase 3: linear head out = h @ Wf + bf --------------------------------
__global__ void k_head(const __bf16* __restrict__ hbf,
                       const __bf16* __restrict__ wf,
                       const float* __restrict__ bfv,
                       float* __restrict__ out) {
  const int wave = blockIdx.x * (blockDim.x >> 5) + (threadIdx.x >> 5); // 0..63
  const int lane = threadIdx.x & 31;
  const int mt = wave >> 4, nt = wave & 15;
  constexpr int KT = Hsz / 32;
  v8f acc = {};
  const __bf16* Arow  = hbf + (size_t)mt * 16 * Hsz;
  const __bf16* Btile = wf  + (size_t)nt * KT * 512;
#pragma unroll 8
  for (int kt = 0; kt < KT; ++kt) {
    v16bf a = load_afrag(Arow + kt * 32, Hsz);
    v16bf b = load_bfrag(Btile + (size_t)kt * 512);
    acc = wmma_bf16(a, b, acc);
  }
  const int nl = lane & 15, mh = (lane >> 4) << 3;
  const int n = nt * 16 + nl;
  const float bv = bfv[n];
#pragma unroll
  for (int v = 0; v < 8; ++v)
    out[(size_t)(mt * 16 + mh + v) * Osz + n] = acc[v] + bv;
}

// ---------------------------------------------------------------------------
extern "C" void kernel_launch(void* const* d_in, const int* in_sizes, int n_in,
                              void* d_out, int out_size, void* d_ws, size_t ws_size,
                              hipStream_t stream) {
  (void)in_sizes; (void)n_in; (void)out_size; (void)ws_size;
  const float* x   = (const float*)d_in[0];
  const float* Wxz = (const float*)d_in[1];
  const float* Whz = (const float*)d_in[2];
  const float* Wxr = (const float*)d_in[3];
  const float* Whr = (const float*)d_in[4];
  const float* Wxh = (const float*)d_in[5];
  const float* Whh = (const float*)d_in[6];
  const float* bz  = (const float*)d_in[7];
  const float* br  = (const float*)d_in[8];
  const float* bh  = (const float*)d_in[9];
  const float* Wf  = (const float*)d_in[10];
  const float* bfv = (const float*)d_in[11];
  float* out = (float*)d_out;

  char* ws = (char*)d_ws;
  __bf16*   G    = (__bf16*)(ws + OFF_G);
  __bf16*   xT   = (__bf16*)(ws + OFF_XT);
  __bf16*   wxz  = (__bf16*)(ws + OFF_WXZ);
  __bf16*   wxr  = (__bf16*)(ws + OFF_WXR);
  __bf16*   wxh  = (__bf16*)(ws + OFF_WXH);
  __bf16*   whz  = (__bf16*)(ws + OFF_WHZ);
  __bf16*   whr  = (__bf16*)(ws + OFF_WHR);
  __bf16*   whh  = (__bf16*)(ws + OFF_WHH);
  __bf16*   wf   = (__bf16*)(ws + OFF_WF);
  float*    h    = (float*)(ws + OFF_H);
  __bf16*   hbf  = (__bf16*)(ws + OFF_HBF);
  float*    zbuf = (float*)(ws + OFF_Z);
  __bf16*   rhbf = (__bf16*)(ws + OFF_RH);
  unsigned* bar  = (unsigned*)(ws + OFF_BAR);

  // 1) conversions / swizzles
  k_swizzle_w<<<(Isz * Hsz + 255) / 256, 256, 0, stream>>>(Wxz, wxz, Isz, Hsz);
  k_swizzle_w<<<(Isz * Hsz + 255) / 256, 256, 0, stream>>>(Wxr, wxr, Isz, Hsz);
  k_swizzle_w<<<(Isz * Hsz + 255) / 256, 256, 0, stream>>>(Wxh, wxh, Isz, Hsz);
  k_swizzle_w<<<(Hsz * Hsz + 255) / 256, 256, 0, stream>>>(Whz, whz, Hsz, Hsz);
  k_swizzle_w<<<(Hsz * Hsz + 255) / 256, 256, 0, stream>>>(Whr, whr, Hsz, Hsz);
  k_swizzle_w<<<(Hsz * Hsz + 255) / 256, 256, 0, stream>>>(Whh, whh, Hsz, Hsz);
  k_swizzle_w<<<(Hsz * Osz + 255) / 256, 256, 0, stream>>>(Wf,  wf,  Hsz, Osz);
  k_convert_x<<<(SB * Isz + 255) / 256, 256, 0, stream>>>(x, xT);
  k_zero<<<(Bsz * Hsz + 255) / 256, 256, 0, stream>>>(h, hbf, bar);

  // 2) precompute input projections: 3 * 2048 * 64 tile-jobs, 8 waves/block
  k_precompute<<<(3 * (SB / 16) * (Hsz / 16)) / 8, 256, 0, stream>>>(
      xT, wxz, wxr, wxh, bz, br, bh, G);

  // 3) persistent recurrence: 64 blocks = 512 waves
  k_recurrent<<<64, 256, 0, stream>>>(G, whz, whr, whh, h, hbf, zbuf, rhbf,
                                      bar, 64u);

  // 4) head
  k_head<<<8, 256, 0, stream>>>(hbf, wf, bfv, out);
}
